// SingleModalAtten_20667382629157
// MI455X (gfx1250) — matrix operand order repaired and checked by reading, not compiled, not run on hardware
//
#include <hip/hip_runtime.h>
#include <math.h>
#include <stdint.h>

// ---------------------------------------------------------------------------
// Problem constants (match reference)
// ---------------------------------------------------------------------------
#define BB    4
#define CC    512
#define TT    2048
#define HH    8
#define CH    64          // C / H
#define GRP   32
#define CG    16          // C / GROUPS
#define K3C   1536        // 3*C
#define SCALE 0.3535533905932738f   // 1/sqrt(sqrt(64))

typedef _Float16 half_t;
typedef __attribute__((ext_vector_type(16))) _Float16 v16h;
typedef __attribute__((ext_vector_type(8)))  float    v8f;
typedef unsigned int u32x4 __attribute__((ext_vector_type(4)));
typedef int          i32x4 __attribute__((ext_vector_type(4)));
typedef int          i32x8 __attribute__((ext_vector_type(8)));

union FragH { v16h v; unsigned int u[8]; };

// ---------------------------------------------------------------------------
// TDM availability
// ---------------------------------------------------------------------------
#ifndef USE_TDM
#if defined(__has_builtin)
#if __has_builtin(__builtin_amdgcn_tensor_load_to_lds)
#define USE_TDM 1
#endif
#endif
#endif
#ifndef USE_TDM
#define USE_TDM 0
#endif

#if USE_TDM
// Issue a 2D TENSOR_LOAD_TO_LDS: tile_d1 rows of tile_d0 contiguous f16
// elements, row stride `stride` elements, into LDS at byte offset
// lds_byte_off.  pad!=0 adds 1 DWORD of LDS padding per 32 DWORDs stored
// (row 64 halves -> effective LDS leading dim 66 halves).
// D# layout per CDNA5 ISA.
__device__ __forceinline__ void tdm_load_2d(unsigned lds_byte_off,
                                            const half_t* gsrc,
                                            unsigned tile_d0, unsigned tile_d1,
                                            unsigned stride, int pad) {
  const unsigned long long ga = (unsigned long long)(uintptr_t)gsrc;
  u32x4 g0;
  g0[0] = 1u;                                        // count=1 (valid), load
  g0[1] = lds_byte_off;                              // lds_addr
  g0[2] = (unsigned)(ga & 0xFFFFFFFFu);              // global_addr[31:0]
  g0[3] = (unsigned)((ga >> 32) & 0x1FFFFFFu)        // global_addr[56:32]
          | (2u << 30);                              // type = 2 ("image")
  unsigned dw0 = (1u << 16);                         // data_size = 2 bytes
  if (pad) dw0 |= (1u << 20) | (4u << 22);           // pad: 1 DWORD / 32 DWORDs
  i32x8 g1;
  g1[0] = (int)dw0;
  g1[1] = (int)((tile_d0 & 0xFFFFu) << 16);          // tensor_dim0 lo
  g1[2] = (int)((tile_d0 >> 16) | ((tile_d1 & 0xFFFFu) << 16)); // dim0 hi|dim1 lo
  g1[3] = (int)((tile_d1 >> 16) | (tile_d0 << 16));  // dim1 hi | tile_dim0
  g1[4] = (int)(tile_d1 & 0xFFFFu);                  // tile_dim1 | tile_dim2=0
  g1[5] = (int)stride;                               // tensor_dim0_stride lo
  g1[6] = 0;                                         // stride hi | dim1_stride lo
  g1[7] = 0;
  const i32x4 z4 = {0, 0, 0, 0};
#if __clang_major__ >= 23
  const i32x8 z8 = {0, 0, 0, 0, 0, 0, 0, 0};
  __builtin_amdgcn_tensor_load_to_lds(g0, g1, z4, z4, z8, 0);
#else
  __builtin_amdgcn_tensor_load_to_lds(g0, g1, z4, z4, 0);
#endif
}
#define LDS_OFF(p) ((unsigned)(uintptr_t)(p))
#else
// Fallback: cooperative copy by all 128 threads (dst_ld in halves)
__device__ __forceinline__ void coop_copy_2d(half_t* dst, const half_t* src,
                                             int rows, int cols, int sstride,
                                             int dld) {
  for (int i = threadIdx.x; i < rows * cols; i += 128) {
    const int r = i / cols, c = i - r * cols;
    dst[r * dld + c] = src[(size_t)r * sstride + c];
  }
}
#endif

// ---------------------------------------------------------------------------
// WMMA fragment helpers (wave32, V_WMMA_F32_16X16X32_F16)
// A frag: M=16 x K=32, src row-major [m][k], even ld.
// B frag: K=32 x N=16, src stored [n][k], even ld.
// C/D f32: lane (l&15)=n; vgpr j is row m = j + (l>>4)*8
// ---------------------------------------------------------------------------
__device__ __forceinline__ v16h load_frag_a(const half_t* base, int ld) {
  const int lane = threadIdx.x & 31;
  const int m  = lane & 15;
  const int hi = lane >> 4;
  FragH f;
#pragma unroll
  for (int j = 0; j < 8; ++j) {
    const int k = ((j & 4) << 2) + (hi << 3) + ((j & 3) << 1);
    f.u[j] = *reinterpret_cast<const unsigned int*>(base + m * ld + k);
  }
  return f.v;
}

__device__ __forceinline__ v16h load_frag_b(const half_t* base, int ld) {
  const int lane = threadIdx.x & 31;
  const int n  = lane & 15;
  const int kb = (lane >> 4) << 3;
  FragH f;
#pragma unroll
  for (int j = 0; j < 8; ++j) {
    f.u[j] = *reinterpret_cast<const unsigned int*>(base + n * ld + kb + ((j & 4) << 2) + ((j & 3) << 1));
  }
  return f.v;
}

__device__ __forceinline__ v8f wmma16(v16h a, v16h b, v8f c) {
  return __builtin_amdgcn_wmma_f32_16x16x32_f16(false, a, false, b,
                                                (short)0, c, false, false);
}

// ---------------------------------------------------------------------------
// Kernel 1: convert f32 weights to f16
// ---------------------------------------------------------------------------
__global__ void convert_w_kernel(const float* __restrict__ qkvw,
                                 const float* __restrict__ projw,
                                 half_t* __restrict__ qkvh,
                                 half_t* __restrict__ projh) {
  const int i = blockIdx.x * 256 + threadIdx.x;
  if (i < K3C * CC)  qkvh[i]  = (half_t)qkvw[i];
  if (i < CC * CC)   projh[i] = (half_t)projw[i];
}

// ---------------------------------------------------------------------------
// Kernel 2: GroupNorm -> f16, written t-major: xn_t[b][t][c]
// ---------------------------------------------------------------------------
__global__ void groupnorm_kernel(const float* __restrict__ x,
                                 const float* __restrict__ nw,
                                 const float* __restrict__ nb,
                                 half_t* __restrict__ xn_t) {
  const int bg = blockIdx.x;            // b*GRP + g
  const int g = bg & (GRP - 1);
  const int b = bg >> 5;
  const size_t base = ((size_t)b * CC + (size_t)g * CG) * TT;
  const int n = CG * TT;                // 32768 contiguous floats

  float s = 0.f, sq = 0.f;
  for (int i = threadIdx.x; i < n; i += 256) {
    const float v = x[base + i];
    s += v; sq += v * v;
  }
  __shared__ float sh1[256], sh2[256];
  sh1[threadIdx.x] = s; sh2[threadIdx.x] = sq;
  __syncthreads();
  for (int off = 128; off > 0; off >>= 1) {
    if (threadIdx.x < off) {
      sh1[threadIdx.x] += sh1[threadIdx.x + off];
      sh2[threadIdx.x] += sh2[threadIdx.x + off];
    }
    __syncthreads();
  }
  const float mean = sh1[0] / (float)n;
  const float var  = sh2[0] / (float)n - mean * mean;
  const float rinv = rsqrtf(var + 1e-5f);

  for (int i = threadIdx.x; i < n; i += 256) {
    const int c = g * CG + (i >> 11);
    const int t = i & (TT - 1);
    const float v = (x[base + i] - mean) * rinv * nw[c] + nb[c];
    xn_t[((size_t)b * TT + t) * CC + c] = (half_t)v;
  }
}

// ---------------------------------------------------------------------------
// Kernel 3: QKV GEMM.  W[o,c] x xn_t[b,t,c]^T.  Block tile 64(o) x 64(t),
// 4 waves x 32x32, K-steps of 32, TDM double-buffered LDS staging.
// Outputs: q_t,k_t t-major [BH][T][64] (scaled); v [BH][64][T].
// ---------------------------------------------------------------------------
__global__ void qkv_gemm_kernel(const half_t* __restrict__ W,    // [1536,512]
                                const half_t* __restrict__ Xt,   // [B,2048,512]
                                const float* __restrict__ bias,  // [1536]
                                half_t* __restrict__ Qt_g,
                                half_t* __restrict__ Kt_g,
                                half_t* __restrict__ Vo) {
  const int b  = blockIdx.z;
  const int m0 = blockIdx.y * 64;
  const int n0 = blockIdx.x * 64;
  const int tid = threadIdx.x, lane = tid & 31, wave = tid >> 5;
  const int wm = (wave >> 1) * 32, wn = (wave & 1) * 32;
  const int hi = lane >> 4, nl = lane & 15;

  __shared__ half_t As[2][64 * 32];     // [m][k]
  __shared__ half_t Bt[2][64 * 32];     // [n][k]
  const half_t* Xb = Xt + (size_t)b * TT * CC;
  const half_t* Wb = W + (size_t)m0 * CC;
  const half_t* Nb = Xb + (size_t)n0 * CC;

  v8f acc[2][2] = {};

#if USE_TDM
  if (wave == 0) {
    tdm_load_2d(LDS_OFF(&As[0][0]), Wb, 32, 64, CC, 0);
    tdm_load_2d(LDS_OFF(&Bt[0][0]), Nb, 32, 64, CC, 0);
  }
#endif
  for (int kki = 0; kki < CC / 32; ++kki) {
    const int cur = kki & 1;
#if USE_TDM
    if (wave == 0) {
      if (kki + 1 < CC / 32) {
        tdm_load_2d(LDS_OFF(&As[cur ^ 1][0]), Wb + (kki + 1) * 32, 32, 64, CC, 0);
        tdm_load_2d(LDS_OFF(&Bt[cur ^ 1][0]), Nb + (kki + 1) * 32, 32, 64, CC, 0);
        __builtin_amdgcn_s_wait_tensorcnt(2);
      } else {
        __builtin_amdgcn_s_wait_tensorcnt(0);
      }
    }
#else
    coop_copy_2d(&As[cur][0], Wb + kki * 32, 64, 32, CC, 32);
    coop_copy_2d(&Bt[cur][0], Nb + kki * 32, 64, 32, CC, 32);
#endif
    __syncthreads();

    const v16h a0 = load_frag_a(&As[cur][wm * 32], 32);
    const v16h a1 = load_frag_a(&As[cur][(wm + 16) * 32], 32);
    const v16h b0 = load_frag_b(&Bt[cur][wn * 32], 32);
    const v16h b1 = load_frag_b(&Bt[cur][(wn + 16) * 32], 32);
    acc[0][0] = wmma16(a0, b0, acc[0][0]);
    acc[0][1] = wmma16(a0, b1, acc[0][1]);
    acc[1][0] = wmma16(a1, b0, acc[1][0]);
    acc[1][1] = wmma16(a1, b1, acc[1][1]);
    __syncthreads();
  }

#pragma unroll
  for (int im = 0; im < 2; ++im)
#pragma unroll
    for (int in = 0; in < 2; ++in)
#pragma unroll
      for (int j = 0; j < 8; ++j) {
        const int o = m0 + wm + im * 16 + j + hi * 8;
        const int t = n0 + wn + in * 16 + nl;
        const float val = acc[im][in][j] + bias[o];
        if (o < CC) {                   // q: [b*8 + o/64][t][o%64], scaled
          const int bh = b * HH + (o >> 6);
          Qt_g[((size_t)bh * TT + t) * CH + (o & 63)] = (half_t)(val * SCALE);
        } else if (o < 2 * CC) {        // k: same layout, scaled
          const int oc = o - CC;
          const int bh = b * HH + (oc >> 6);
          Kt_g[((size_t)bh * TT + t) * CH + (oc & 63)] = (half_t)(val * SCALE);
        } else {                        // v: [bh][c][t]
          const int oc = o - 2 * CC;
          Vo[((size_t)b * CC + oc) * TT + t] = (half_t)val;
        }
      }
}

// ---------------------------------------------------------------------------
// Kernel 4: flash attention.  One block = one head x 64 queries; wave w owns
// query rows [w*16, w*16+16).  TDM stages Q (once) and double-buffered K/V
// tiles; P round-trips through per-wave LDS rows to become the A operand.
// Output written t-major: a_t[b][t][C].
// ---------------------------------------------------------------------------
__global__ void attn_kernel(const half_t* __restrict__ Q,   // [BH][T][64]
                            const half_t* __restrict__ K,   // [BH][T][64]
                            const half_t* __restrict__ V,   // [BH][64][T]
                            half_t* __restrict__ At) {      // [B][T][C]
  const int bh = blockIdx.y;
  const int t0 = blockIdx.x * 64;
  const int tid = threadIdx.x, lane = tid & 31, wave = tid >> 5;
  const int hi = lane >> 4, nl = lane & 15;
  const half_t* Qh = Q + ((size_t)bh * TT) * CH;
  const half_t* Kh = K + ((size_t)bh * TT) * CH;
  const half_t* Vh = V + (size_t)bh * CH * TT;

  __shared__ half_t Qt[64 * 66];        // [t][c]
  __shared__ half_t Kt[2][64 * 66];     // [s][c]
  __shared__ half_t Vl[2][64 * 66];     // [c][s]
  __shared__ half_t Ps[64 * 66];        // [t][s], per-wave 16-row regions

#if USE_TDM
  if (wave == 0) {
    tdm_load_2d(LDS_OFF(&Qt[0]),    Qh + (size_t)t0 * CH, 64, 64, CH, 1);
    tdm_load_2d(LDS_OFF(&Kt[0][0]), Kh,                   64, 64, CH, 1);
    tdm_load_2d(LDS_OFF(&Vl[0][0]), Vh,                   64, 64, TT, 1);
  }
#else
  coop_copy_2d(&Qt[0], Qh + (size_t)t0 * CH, 64, 64, CH, 66);
  __syncthreads();
#endif

  v16h qa0 = {}, qa1 = {};
  float rm[8], rs[8];
#pragma unroll
  for (int j = 0; j < 8; ++j) { rm[j] = -1e30f; rs[j] = 0.f; }
  v8f oacc[4] = {};

  for (int st = 0; st < TT; st += 64) {
    const int cur = (st >> 6) & 1;
#if USE_TDM
    if (wave == 0) {
      if (st + 64 < TT) {
        tdm_load_2d(LDS_OFF(&Kt[cur ^ 1][0]), Kh + (size_t)(st + 64) * CH, 64, 64, CH, 1);
        tdm_load_2d(LDS_OFF(&Vl[cur ^ 1][0]), Vh + st + 64,                64, 64, TT, 1);
        __builtin_amdgcn_s_wait_tensorcnt(2);
      } else {
        __builtin_amdgcn_s_wait_tensorcnt(0);
      }
    }
#else
    coop_copy_2d(&Kt[cur][0], Kh + (size_t)st * CH, 64, 64, CH, 66);
    coop_copy_2d(&Vl[cur][0], Vh + st,              64, 64, TT, 66);
#endif
    __syncthreads();
    if (st == 0) {
      qa0 = load_frag_a(&Qt[(wave * 16) * 66 + 0], 66);
      qa1 = load_frag_a(&Qt[(wave * 16) * 66 + 32], 66);
    }

    // S strip [16 x 64] for this wave's query rows (K-dim = ch = 64)
    v8f sacc[4];
#pragma unroll
    for (int sf = 0; sf < 4; ++sf) {
      const v16h bk0 = load_frag_b(&Kt[cur][(sf * 16) * 66 + 0], 66);
      const v16h bk1 = load_frag_b(&Kt[cur][(sf * 16) * 66 + 32], 66);
      v8f z = {};
      z = wmma16(qa0, bk0, z);
      sacc[sf] = wmma16(qa1, bk1, z);
    }

    // online softmax; row m = j + hi*8 lives in 16 lanes of this half-wave
#pragma unroll
    for (int j = 0; j < 8; ++j) {
      float mx = fmaxf(fmaxf(sacc[0][j], sacc[1][j]),
                       fmaxf(sacc[2][j], sacc[3][j]));
      mx = fmaxf(mx, __shfl_xor(mx, 1, 32));
      mx = fmaxf(mx, __shfl_xor(mx, 2, 32));
      mx = fmaxf(mx, __shfl_xor(mx, 4, 32));
      mx = fmaxf(mx, __shfl_xor(mx, 8, 32));
      const float mnew = fmaxf(rm[j], mx);
      const float corr = __expf(rm[j] - mnew);
      float psum = 0.f;
#pragma unroll
      for (int sf = 0; sf < 4; ++sf) {
        const float p = __expf(sacc[sf][j] - mnew);
        sacc[sf][j] = p;
        psum += p;
      }
      psum += __shfl_xor(psum, 1, 32);
      psum += __shfl_xor(psum, 2, 32);
      psum += __shfl_xor(psum, 4, 32);
      psum += __shfl_xor(psum, 8, 32);
      rs[j] = rs[j] * corr + psum;
      rm[j] = mnew;
#pragma unroll
      for (int cf = 0; cf < 4; ++cf) oacc[cf][j] *= corr;
    }

    // spill P (f16) into this wave's LDS rows, reload as A fragments
#pragma unroll
    for (int sf = 0; sf < 4; ++sf)
#pragma unroll
      for (int j = 0; j < 8; ++j)
        Ps[(wave * 16 + j + hi * 8) * 66 + sf * 16 + nl] = (half_t)sacc[sf][j];

    const v16h pa0 = load_frag_a(&Ps[(wave * 16) * 66 + 0], 66);
    const v16h pa1 = load_frag_a(&Ps[(wave * 16) * 66 + 32], 66);
#pragma unroll
    for (int cf = 0; cf < 4; ++cf) {
      const v16h vb0 = load_frag_b(&Vl[cur][(cf * 16) * 66 + 0], 66);
      const v16h vb1 = load_frag_b(&Vl[cur][(cf * 16) * 66 + 32], 66);
      oacc[cf] = wmma16(pa0, vb0, oacc[cf]);
      oacc[cf] = wmma16(pa1, vb1, oacc[cf]);
    }
    __syncthreads();
  }

  // normalize; write a t-major: At[b][t][h*64 + c]
  const int b = bh >> 3;
  const int hcol = (bh & 7) * CH;
#pragma unroll
  for (int cf = 0; cf < 4; ++cf)
#pragma unroll
    for (int j = 0; j < 8; ++j) {
      const int c = cf * 16 + nl;
      const int t = t0 + wave * 16 + j + hi * 8;
      const float val = oacc[cf][j] / rs[j];
      At[((size_t)b * TT + t) * CC + hcol + c] = (half_t)val;
    }
}

// ---------------------------------------------------------------------------
// Kernel 5: output projection + bias + residual (f32 out)
// ---------------------------------------------------------------------------
__global__ void proj_gemm_kernel(const half_t* __restrict__ W,   // [512,512]
                                 const half_t* __restrict__ Xt,  // a_t [B,2048,512]
                                 const float* __restrict__ bias, // [512]
                                 const float* __restrict__ xres, // [B,512,2048]
                                 float* __restrict__ out) {
  const int b  = blockIdx.z;
  const int m0 = blockIdx.y * 64;
  const int n0 = blockIdx.x * 64;
  const int tid = threadIdx.x, lane = tid & 31, wave = tid >> 5;
  const int wm = (wave >> 1) * 32, wn = (wave & 1) * 32;
  const int hi = lane >> 4, nl = lane & 15;

  __shared__ half_t As[2][64 * 32];
  __shared__ half_t Bt[2][64 * 32];
  const half_t* Wb = W + (size_t)m0 * CC;
  const half_t* Nb = Xt + ((size_t)b * TT + n0) * CC;

  v8f acc[2][2] = {};

#if USE_TDM
  if (wave == 0) {
    tdm_load_2d(LDS_OFF(&As[0][0]), Wb, 32, 64, CC, 0);
    tdm_load_2d(LDS_OFF(&Bt[0][0]), Nb, 32, 64, CC, 0);
  }
#endif
  for (int kki = 0; kki < CC / 32; ++kki) {
    const int cur = kki & 1;
#if USE_TDM
    if (wave == 0) {
      if (kki + 1 < CC / 32) {
        tdm_load_2d(LDS_OFF(&As[cur ^ 1][0]), Wb + (kki + 1) * 32, 32, 64, CC, 0);
        tdm_load_2d(LDS_OFF(&Bt[cur ^ 1][0]), Nb + (kki + 1) * 32, 32, 64, CC, 0);
        __builtin_amdgcn_s_wait_tensorcnt(2);
      } else {
        __builtin_amdgcn_s_wait_tensorcnt(0);
      }
    }
#else
    coop_copy_2d(&As[cur][0], Wb + kki * 32, 64, 32, CC, 32);
    coop_copy_2d(&Bt[cur][0], Nb + kki * 32, 64, 32, CC, 32);
#endif
    __syncthreads();

    const v16h a0 = load_frag_a(&As[cur][wm * 32], 32);
    const v16h a1 = load_frag_a(&As[cur][(wm + 16) * 32], 32);
    const v16h b0 = load_frag_b(&Bt[cur][wn * 32], 32);
    const v16h b1 = load_frag_b(&Bt[cur][(wn + 16) * 32], 32);
    acc[0][0] = wmma16(a0, b0, acc[0][0]);
    acc[0][1] = wmma16(a0, b1, acc[0][1]);
    acc[1][0] = wmma16(a1, b0, acc[1][0]);
    acc[1][1] = wmma16(a1, b1, acc[1][1]);
    __syncthreads();
  }

#pragma unroll
  for (int im = 0; im < 2; ++im)
#pragma unroll
    for (int in = 0; in < 2; ++in)
#pragma unroll
      for (int j = 0; j < 8; ++j) {
        const int m = m0 + wm + im * 16 + j + hi * 8;
        const int t = n0 + wn + in * 16 + nl;
        const size_t idx = ((size_t)b * CC + m) * TT + t;
        out[idx] = xres[idx] + bias[m] + acc[im][in][j];
      }
}

// ---------------------------------------------------------------------------
// Host-side launch
// ---------------------------------------------------------------------------
extern "C" void kernel_launch(void* const* d_in, const int* in_sizes, int n_in,
                              void* d_out, int out_size, void* d_ws, size_t ws_size,
                              hipStream_t stream) {
  (void)in_sizes; (void)n_in; (void)out_size; (void)ws_size;

  const float* x      = (const float*)d_in[0];
  const float* norm_w = (const float*)d_in[1];
  const float* norm_b = (const float*)d_in[2];
  const float* qkv_w  = (const float*)d_in[3];
  const float* qkv_b  = (const float*)d_in[4];
  const float* proj_w = (const float*)d_in[5];
  const float* proj_b = (const float*)d_in[6];
  float* out = (float*)d_out;

  // workspace layout (f16 elements)
  half_t* ws = (half_t*)d_ws;
  const size_t BCT = (size_t)BB * CC * TT;            // 4,194,304
  half_t* xn_t     = ws;                              // [B][T][C]
  half_t* qkv_w_h  = xn_t + BCT;
  half_t* proj_w_h = qkv_w_h + (size_t)K3C * CC;
  half_t* q_t      = proj_w_h + (size_t)CC * CC;      // [BH][T][64]
  half_t* k_t      = q_t + BCT;                       // [BH][T][64]
  half_t* v_h      = k_t + BCT;                       // [BH][64][T]
  half_t* a_t      = v_h + BCT;                       // [B][T][C]

  convert_w_kernel<<<(K3C * CC + 255) / 256, 256, 0, stream>>>(
      qkv_w, proj_w, qkv_w_h, proj_w_h);

  groupnorm_kernel<<<BB * GRP, 256, 0, stream>>>(x, norm_w, norm_b, xn_t);

  qkv_gemm_kernel<<<dim3(TT / 64, K3C / 64, BB), 128, 0, stream>>>(
      qkv_w_h, xn_t, qkv_b, q_t, k_t, v_h);

  attn_kernel<<<dim3(TT / 64, BB * HH), 128, 0, stream>>>(q_t, k_t, v_h, a_t);

  proj_gemm_kernel<<<dim3(TT / 64, CC / 64, BB), 128, 0, stream>>>(
      proj_w_h, a_t, proj_b, x, out);
}
